// BigBirdForQuestionAnswering_68942815035638
// MI455X (gfx1250) — matrix operand order repaired and checked by reading, not compiled, not run on hardware
//
#include <hip/hip_runtime.h>
#include <hip/hip_bf16.h>
#include <math.h>

// ---------------- constants ----------------
constexpr int B_ = 2, S_ = 4096, D_ = 768, L_ = 12, H_ = 12, HD_ = 64, F_ = 3072;
constexpr int BS_ = 64, NB_ = S_ / BS_, R_ = 3;
constexpr float NEGV = -1000000000.0f;
constexpr float SCALE = 0.125f; // 1/sqrt(64)

// ---------------- WMMA types ----------------
typedef __attribute__((ext_vector_type(16))) __bf16 v16bf;
typedef __attribute__((ext_vector_type(8)))  float  v8f;
typedef __attribute__((ext_vector_type(4)))  unsigned int u32x4;

union ABf { v16bf v; unsigned short s[16]; };

__device__ __forceinline__ unsigned short f2bf(float f) {
    union { float f; unsigned u; } c; c.f = f;
    unsigned u = c.u;
    u += 0x7FFFu + ((u >> 16) & 1u);       // round-to-nearest-even
    return (unsigned short)(u >> 16);
}
__device__ __forceinline__ float bf2f(unsigned short h) {
    union { unsigned u; float f; } c; c.u = ((unsigned)h) << 16;
    return c.f;
}
__device__ __forceinline__ float gelu_tanh(float x) {
    float x3 = x * x * x;
    float t = tanhf(0.7978845608028654f * (x + 0.044715f * x3));
    return 0.5f * x * (1.0f + t);
}

// ---- CDNA5 async global->LDS copy (ASYNCcnt path) ----
__device__ __forceinline__ void async_ld_b128(void* lds, const void* gsrc) {
    unsigned l = (unsigned)(unsigned long long)lds;              // low 32 bits = LDS offset
    unsigned long long g = (unsigned long long)gsrc;
    asm volatile("global_load_async_to_lds_b128 %0, %1, off" :: "v"(l), "v"(g) : "memory");
}
#define WAIT_ASYNC_2() asm volatile("s_wait_asynccnt 0x2" ::: "memory")
#define WAIT_ASYNC_0() asm volatile("s_wait_asynccnt 0x0" ::: "memory")

// Load the two contiguous 16B halves of an A operand (K = half*8+0..7, 16+half*8+0..7)
__device__ __forceinline__ void loadA(ABf& a, const unsigned short* rowBase, int half) {
    const unsigned short* p = rowBase + half * 8;
    *(u32x4*)&a.s[0] = *(const u32x4*)p;
    *(u32x4*)&a.s[8] = *(const u32x4*)(p + 16);
}
// Load the two contiguous 16B halves of a B operand (K = half*8+0..7, 16+half*8+0..7)
__device__ __forceinline__ void loadB(ABf& b, const unsigned short* rowBase, int half) {
    const unsigned short* p = rowBase + half * 8;
    *(u32x4*)&b.s[0] = *(const u32x4*)p;
    *(u32x4*)&b.s[8] = *(const u32x4*)(p + 16);
}

// ---------------- weight convert + transpose: W[K x N] f32 -> WT[N x K] bf16 ----------------
__global__ __launch_bounds__(256) void conv_transpose(const float* __restrict__ W,
                                                      unsigned short* __restrict__ WT,
                                                      int K, int N) {
    __shared__ unsigned short t[32][33];
    int l = blockIdx.z;
    const float* Wl = W + (size_t)l * K * N;
    unsigned short* WTl = WT + (size_t)l * K * N;
    int tn = blockIdx.x * 32, tk = blockIdx.y * 32;
    int tid = threadIdx.x;
    #pragma unroll
    for (int i = 0; i < 4; i++) {
        int e = tid + i * 256; int r = e >> 5, c = e & 31;
        t[r][c] = f2bf(Wl[(size_t)(tk + r) * N + tn + c]);
    }
    __syncthreads();
    #pragma unroll
    for (int i = 0; i < 4; i++) {
        int e = tid + i * 256; int r = e >> 5, c = e & 31;
        WTl[(size_t)(tn + r) * K + tk + c] = t[c][r];
    }
}

// ---------------- idx / keep precompute ----------------
__global__ void idx_kernel(const int* __restrict__ rb, int* __restrict__ idxb, int* __restrict__ keepb) {
    int t = blockIdx.x * blockDim.x + threadIdx.x;
    if (t >= L_ * NB_) return;
    int l = t / NB_, n = t % NB_;
    int id[8];
    id[0] = 0; id[1] = NB_ - 1;
    id[2] = (n - 1 < 0) ? 0 : n - 1;
    id[3] = n;
    id[4] = (n + 1 > NB_ - 1) ? NB_ - 1 : n + 1;
    for (int r = 0; r < R_; r++) id[5 + r] = rb[(l * NB_ + n) * R_ + r];
    for (int j = 0; j < 8; j++) {
        int kp = 1;
        for (int jj = 0; jj < j; jj++) if (id[jj] == id[j]) kp = 0;
        idxb[t * 8 + j] = id[j];
        keepb[t * 8 + j] = kp;
    }
}

// ---------------- embedding + layernorm (f32 + bf16 outputs) ----------------
__global__ __launch_bounds__(256) void embed_ln(const int* __restrict__ ids,
                                                const float* __restrict__ wemb,
                                                const float* __restrict__ pemb,
                                                const float* __restrict__ g,
                                                const float* __restrict__ be,
                                                float* __restrict__ x,
                                                unsigned short* __restrict__ xb) {
    int tok = blockIdx.x;
    int s = tok % S_;
    int id = ids[tok];
    __shared__ float red[256];
    float e[3];
    #pragma unroll
    for (int i = 0; i < 3; i++) {
        int c = threadIdx.x + i * 256;
        e[i] = wemb[(size_t)id * D_ + c] + pemb[(size_t)s * D_ + c];
    }
    float sum = e[0] + e[1] + e[2];
    red[threadIdx.x] = sum; __syncthreads();
    for (int st = 128; st > 0; st >>= 1) { if (threadIdx.x < st) red[threadIdx.x] += red[threadIdx.x + st]; __syncthreads(); }
    float mean = red[0] / D_;
    __syncthreads();
    float vs = 0.f;
    #pragma unroll
    for (int i = 0; i < 3; i++) { float d = e[i] - mean; vs += d * d; }
    red[threadIdx.x] = vs; __syncthreads();
    for (int st = 128; st > 0; st >>= 1) { if (threadIdx.x < st) red[threadIdx.x] += red[threadIdx.x + st]; __syncthreads(); }
    float inv = rsqrtf(red[0] / D_ + 1e-12f);
    #pragma unroll
    for (int i = 0; i < 3; i++) {
        int c = threadIdx.x + i * 256;
        float o = (e[i] - mean) * inv * g[c] + be[c];
        x[(size_t)tok * D_ + c] = o;
        xb[(size_t)tok * D_ + c] = f2bf(o);
    }
}

// ---------------- residual add + layernorm (f32 + bf16 outputs) ----------------
__global__ __launch_bounds__(256) void add_ln(const float* __restrict__ a,
                                              const float* __restrict__ b,
                                              const float* __restrict__ g,
                                              const float* __restrict__ be,
                                              float* __restrict__ out,
                                              unsigned short* __restrict__ outb) {
    int tok = blockIdx.x;
    __shared__ float red[256];
    float e[3];
    #pragma unroll
    for (int i = 0; i < 3; i++) {
        int c = threadIdx.x + i * 256;
        e[i] = a[(size_t)tok * D_ + c] + b[(size_t)tok * D_ + c];
    }
    float sum = e[0] + e[1] + e[2];
    red[threadIdx.x] = sum; __syncthreads();
    for (int st = 128; st > 0; st >>= 1) { if (threadIdx.x < st) red[threadIdx.x] += red[threadIdx.x + st]; __syncthreads(); }
    float mean = red[0] / D_;
    __syncthreads();
    float vs = 0.f;
    #pragma unroll
    for (int i = 0; i < 3; i++) { float d = e[i] - mean; vs += d * d; }
    red[threadIdx.x] = vs; __syncthreads();
    for (int st = 128; st > 0; st >>= 1) { if (threadIdx.x < st) red[threadIdx.x] += red[threadIdx.x + st]; __syncthreads(); }
    float inv = rsqrtf(red[0] / D_ + 1e-12f);
    #pragma unroll
    for (int i = 0; i < 3; i++) {
        int c = threadIdx.x + i * 256;
        float o = (e[i] - mean) * inv * g[c] + be[c];
        out[(size_t)tok * D_ + c] = o;
        outb[(size_t)tok * D_ + c] = f2bf(o);
    }
}

// ---------------- WMMA GEMM (bf16 A * bf16 W^T) ----------------
// Triple-buffered async staging, one barrier per k-step, peeled tail.
// C = act(A[MxK] * W[KxN] + bias); WT is W transposed: [N x K].
__global__ __launch_bounds__(256) void gemm_bf16(const unsigned short* __restrict__ A,
                                                 const unsigned short* __restrict__ WT,
                                                 const float* __restrict__ bias,
                                                 float* __restrict__ Cf,
                                                 unsigned short* __restrict__ Cb,
                                                 int M, int N, int K, int act) {
    __shared__ alignas(16) unsigned short As[3][64][40];
    __shared__ alignas(16) unsigned short Bs[3][64][40];
    const int tid = threadIdx.x, lane = tid & 31, wave = tid >> 5;
    const int half = lane >> 4, ln = lane & 15;
    const int m0 = blockIdx.y * 64, n0 = blockIdx.x * 64;
    const int subM = (wave >> 1) * 16, subN0 = (wave & 1) * 32;
    const int sr = tid >> 2, sc = (tid & 3) * 8;   // staging: one 16B chunk each of A and B tile
    v8f acc0 = {}, acc1 = {};
    const int T = K >> 5;
    const unsigned short* Arow = A  + (size_t)(m0 + sr) * K + sc;
    const unsigned short* Brow = WT + (size_t)(n0 + sr) * K + sc;

    auto stage = [&](int kt) {
        int bs = kt % 3;
        async_ld_b128(&As[bs][sr][sc], Arow + kt * 32);
        async_ld_b128(&Bs[bs][sr][sc], Brow + kt * 32);
    };
    auto compute = [&](int kt) {
        int cs = kt % 3;
        ABf a, b0, b1;
        loadA(a,  &As[cs][subM + ln][0], half);
        loadB(b0, &Bs[cs][subN0 + ln][0], half);
        loadB(b1, &Bs[cs][subN0 + 16 + ln][0], half);
        acc0 = __builtin_amdgcn_wmma_f32_16x16x32_bf16(false, a.v, false, b0.v, (short)0, acc0, false, false);
        acc1 = __builtin_amdgcn_wmma_f32_16x16x32_bf16(false, a.v, false, b1.v, (short)0, acc1, false, false);
    };

    stage(0);
    stage(1);
    for (int kt = 0; kt < T - 2; kt++) {
        WAIT_ASYNC_2();            // tile kt landed; tile kt+1 still in flight
        __syncthreads();
        compute(kt);
        stage(kt + 2);             // safe: buffer (kt+2)%3 last read at compute(kt-1), fenced by this barrier
    }
    WAIT_ASYNC_2();
    __syncthreads();
    compute(T - 2);
    WAIT_ASYNC_0();
    __syncthreads();
    compute(T - 1);

    #pragma unroll
    for (int r = 0; r < 8; r++) {
        int row = m0 + subM + half * 8 + r;
        int c0 = n0 + subN0 + ln;
        int c1 = c0 + 16;
        float v0 = acc0[r] + bias[c0];
        float v1 = acc1[r] + bias[c1];
        if (act) { v0 = gelu_tanh(v0); v1 = gelu_tanh(v1); }
        if (Cf) { Cf[(size_t)row * N + c0] = v0; Cf[(size_t)row * N + c1] = v1; }
        if (Cb) { Cb[(size_t)row * N + c0] = f2bf(v0); Cb[(size_t)row * N + c1] = f2bf(v1); }
    }
}

// ---------------- block-sparse attention (non-global blocks), bf16 in/out ----------------
__global__ __launch_bounds__(256) void sparse_attn(const unsigned short* __restrict__ q,
                                                   const unsigned short* __restrict__ k,
                                                   const unsigned short* __restrict__ v,
                                                   const int* __restrict__ am,
                                                   const int* __restrict__ idxb,
                                                   const int* __restrict__ keepb,
                                                   unsigned short* __restrict__ out,
                                                   int layer) {
    __shared__ alignas(16) unsigned short Qs[64][72];
    __shared__ alignas(16) unsigned short KVpool[512 * 72];   // K as [key][72]; later V^T as [hd][520]
    __shared__ alignas(16) unsigned short Sb[64][520];        // scores then probabilities (bf16)
    __shared__ int sIdx[8];
    __shared__ int sKeep[8];
    int gid = blockIdx.x;
    int n = 1 + gid % (NB_ - 2); int t2 = gid / (NB_ - 2);
    int h = t2 % H_; int b = t2 / H_;
    const int tid = threadIdx.x, lane = tid & 31, wave = tid >> 5;
    const int half = lane >> 4, ln = lane & 15;
    if (tid < 8) {
        int o = (layer * NB_ + n) * 8 + tid;
        sIdx[tid] = idxb[o];
        sKeep[tid] = keepb[o];
    }
    __syncthreads();
    // async stage Q (64x64) : 512 chunks
    #pragma unroll
    for (int i = 0; i < 2; i++) {
        int e = tid + i * 256; int r = e >> 3, c = (e & 7) * 8;
        async_ld_b128(&Qs[r][c], q + (size_t)(b * S_ + n * BS_ + r) * D_ + h * HD_ + c);
    }
    // async stage gathered K (512x64) : 4096 chunks
    for (int i = 0; i < 16; i++) {
        int e = tid + i * 256; int r = e >> 3, c = (e & 7) * 8;
        int blk = sIdx[r >> 6];
        async_ld_b128(&KVpool[r * 72 + c], k + (size_t)(b * S_ + blk * BS_ + (r & 63)) * D_ + h * HD_ + c);
    }
    WAIT_ASYNC_0();
    __syncthreads();
    // scores S(64x512) = Q K^T * scale, masked -> Sb (bf16)
    for (int tt = 0; tt < 16; tt++) {
        int tile = wave * 16 + tt; int mi = tile >> 5, ni = tile & 31;
        v8f acc = {};
        #pragma unroll
        for (int c0 = 0; c0 < 64; c0 += 32) {
            ABf a, bm;
            loadA(a, &Qs[mi * 16 + ln][c0], half);
            loadB(bm, &KVpool[(ni * 16 + ln) * 72 + c0], half);
            acc = __builtin_amdgcn_wmma_f32_16x16x32_bf16(false, a.v, false, bm.v, (short)0, acc, false, false);
        }
        int col = ni * 16 + ln;
        int slot = col >> 6, pos = col & 63;
        int mok = sKeep[slot] && (am[(size_t)b * S_ + sIdx[slot] * BS_ + pos] > 0);
        #pragma unroll
        for (int r = 0; r < 8; r++) {
            int row = mi * 16 + half * 8 + r;
            Sb[row][col] = f2bf(mok ? acc[r] * SCALE : NEGV);
        }
    }
    __syncthreads();
    // softmax: 4 threads per row, wave-local shuffles for the reductions
    {
        int row = tid >> 2, q4 = tid & 3;
        float mx = -1e30f;
        for (int c = q4; c < 512; c += 4) mx = fmaxf(mx, bf2f(Sb[row][c]));
        mx = fmaxf(mx, __shfl_xor(mx, 1, 32));
        mx = fmaxf(mx, __shfl_xor(mx, 2, 32));
        float sum = 0.f;
        for (int c = q4; c < 512; c += 4) {
            float e = __expf(bf2f(Sb[row][c]) - mx);
            sum += e;
            Sb[row][c] = f2bf(e);
        }
        sum += __shfl_xor(sum, 1, 32);
        sum += __shfl_xor(sum, 2, 32);
        float inv = 1.f / sum;
        for (int c = q4; c < 512; c += 4) Sb[row][c] = f2bf(bf2f(Sb[row][c]) * inv);
    }
    __syncthreads();
    // build V^T (64 x 512) into KVpool: coalesced global read, bank-skewed LDS scatter
    for (int i = 0; i < 128; i++) {
        int e = tid + i * 256; int key = e >> 6, hd = e & 63;
        int blk = sIdx[key >> 6];
        KVpool[hd * 520 + key] = v[(size_t)(b * S_ + blk * BS_ + (key & 63)) * D_ + h * HD_ + hd];
    }
    __syncthreads();
    // O(64x64) = P(64x512) * V(512x64)
    for (int tt = 0; tt < 2; tt++) {
        int tile = wave * 2 + tt; int mi = tile >> 2, ni = tile & 3;
        v8f acc = {};
        for (int c0 = 0; c0 < 512; c0 += 32) {
            ABf a, bm;
            loadA(a, &Sb[mi * 16 + ln][c0], half);
            loadB(bm, &KVpool[(ni * 16 + ln) * 520 + c0], half);
            acc = __builtin_amdgcn_wmma_f32_16x16x32_bf16(false, a.v, false, bm.v, (short)0, acc, false, false);
        }
        #pragma unroll
        for (int r = 0; r < 8; r++) {
            int row = mi * 16 + half * 8 + r, col = ni * 16 + ln;
            out[(size_t)(b * S_ + n * BS_ + row) * D_ + h * HD_ + col] = f2bf(acc[r]);
        }
    }
}

// ---------------- global attention for blocks 0 and NB-1 (flash style), bf16 in/out ----------------
__global__ __launch_bounds__(256) void global_attn(const unsigned short* __restrict__ q,
                                                   const unsigned short* __restrict__ k,
                                                   const unsigned short* __restrict__ v,
                                                   const int* __restrict__ am,
                                                   unsigned short* __restrict__ out) {
    __shared__ alignas(16) unsigned short Qg[128][72];
    __shared__ alignas(16) unsigned short Kt[64][72];
    __shared__ alignas(16) unsigned short VT[64][72];   // V^T: [hd][key]
    __shared__ float St[128][72];
    __shared__ alignas(16) unsigned short Pt[128][72];
    __shared__ float Oa[128][72];
    __shared__ float mrow[128], lrow[128];
    int h = blockIdx.x % H_, b = blockIdx.x / H_;
    const int tid = threadIdx.x, lane = tid & 31, wave = tid >> 5;
    const int half = lane >> 4, ln = lane & 15;
    // async stage Qg (128x64)
    #pragma unroll
    for (int i = 0; i < 4; i++) {
        int e = tid + i * 256; int r = e >> 3, c = (e & 7) * 8;
        int srow = (r < 64) ? r : ((NB_ - 1) * BS_ + (r - 64));
        async_ld_b128(&Qg[r][c], q + (size_t)(b * S_ + srow) * D_ + h * HD_ + c);
    }
    #pragma unroll
    for (int i = 0; i < 32; i++) {
        int e = tid + i * 256; int r = e >> 6, c = e & 63;
        Oa[r][c] = 0.f;
    }
    for (int i = tid; i < 128; i += 256) { mrow[i] = -1e30f; lrow[i] = 0.f; }
    WAIT_ASYNC_0();
    __syncthreads();
    for (int kt = 0; kt < NB_; kt++) {
        // async stage K tile; manual-transpose stage V tile
        #pragma unroll
        for (int i = 0; i < 2; i++) {
            int e = tid + i * 256; int r = e >> 3, c = (e & 7) * 8;
            async_ld_b128(&Kt[r][c], k + (size_t)(b * S_ + kt * BS_ + r) * D_ + h * HD_ + c);
        }
        #pragma unroll
        for (int i = 0; i < 16; i++) {
            int e = tid + i * 256; int key = e >> 6, hd = e & 63;
            VT[hd][key] = v[(size_t)(b * S_ + kt * BS_ + key) * D_ + h * HD_ + hd];
        }
        WAIT_ASYNC_0();
        __syncthreads();
        // S_tile(128x64) = Qg * Kt^T * scale
        for (int tt = 0; tt < 4; tt++) {
            int tile = wave * 4 + tt; int mi = tile >> 2, ni = tile & 3;
            v8f acc = {};
            #pragma unroll
            for (int c0 = 0; c0 < 64; c0 += 32) {
                ABf a, bm;
                loadA(a, &Qg[mi * 16 + ln][c0], half);
                loadB(bm, &Kt[ni * 16 + ln][c0], half);
                acc = __builtin_amdgcn_wmma_f32_16x16x32_bf16(false, a.v, false, bm.v, (short)0, acc, false, false);
            }
            int col = ni * 16 + ln;
            int mok = am[(size_t)b * S_ + kt * BS_ + col] > 0;
            #pragma unroll
            for (int r = 0; r < 8; r++)
                St[mi * 16 + half * 8 + r][col] = mok ? acc[r] * SCALE : NEGV;
        }
        __syncthreads();
        // flash rescale: 2 threads per row (lanes 2k,2k+1 of one wave)
        {
            int row = tid >> 1, hf = tid & 1;
            int cbase = hf * 32;
            float mold = mrow[row];
            float lmx = -1e30f;
            for (int c = cbase; c < cbase + 32; c++) lmx = fmaxf(lmx, St[row][c]);
            lmx = fmaxf(lmx, __shfl_xor(lmx, 1, 32));
            float newm = fmaxf(mold, lmx);
            float corr = __expf(mold - newm);
            float s = 0.f;
            for (int c = cbase; c < cbase + 32; c++) {
                float e = __expf(St[row][c] - newm);
                s += e;
                Pt[row][c] = f2bf(e);
            }
            s += __shfl_xor(s, 1, 32);
            if (hf == 0) { lrow[row] = lrow[row] * corr + s; mrow[row] = newm; }
            for (int c = cbase; c < cbase + 32; c++) Oa[row][c] *= corr;
        }
        __syncthreads();
        // Oa += Pt * Vt
        for (int tt = 0; tt < 4; tt++) {
            int tile = wave * 4 + tt; int mi = tile >> 2, ni = tile & 3;
            int col = ni * 16 + ln;
            v8f acc;
            #pragma unroll
            for (int r = 0; r < 8; r++) acc[r] = Oa[mi * 16 + half * 8 + r][col];
            #pragma unroll
            for (int c0 = 0; c0 < 64; c0 += 32) {
                ABf a, bm;
                loadA(a, &Pt[mi * 16 + ln][c0], half);
                loadB(bm, &VT[col][c0], half);
                acc = __builtin_amdgcn_wmma_f32_16x16x32_bf16(false, a.v, false, bm.v, (short)0, acc, false, false);
            }
            #pragma unroll
            for (int r = 0; r < 8; r++) Oa[mi * 16 + half * 8 + r][col] = acc[r];
        }
        __syncthreads();
    }
    if (tid < 128) {
        float inv = 1.f / lrow[tid];
        int srow = (tid < 64) ? tid : ((NB_ - 1) * BS_ + (tid - 64));
        for (int c = 0; c < 64; c++)
            out[(size_t)(b * S_ + srow) * D_ + h * HD_ + c] = f2bf(Oa[tid][c] * inv);
    }
}

// ---------------- QA head: logits = x @ Wqa + bqa ----------------
__global__ __launch_bounds__(256) void qa_proj(const float* __restrict__ x,
                                               const float* __restrict__ Wqa,
                                               const float* __restrict__ bqa,
                                               float* __restrict__ out) {
    int tok = blockIdx.x;
    __shared__ float r0[256], r1[256];
    float s0 = 0.f, s1 = 0.f;
    for (int c = threadIdx.x; c < D_; c += 256) {
        float xv = x[(size_t)tok * D_ + c];
        s0 += xv * Wqa[c * 2 + 0];
        s1 += xv * Wqa[c * 2 + 1];
    }
    r0[threadIdx.x] = s0; r1[threadIdx.x] = s1; __syncthreads();
    for (int st = 128; st > 0; st >>= 1) {
        if (threadIdx.x < st) { r0[threadIdx.x] += r0[threadIdx.x + st]; r1[threadIdx.x] += r1[threadIdx.x + st]; }
        __syncthreads();
    }
    if (threadIdx.x == 0) {
        out[tok] = r0[0] + bqa[0];
        out[(size_t)B_ * S_ + tok] = r1[0] + bqa[1];
    }
}

// ---------------- host orchestration ----------------
extern "C" void kernel_launch(void* const* d_in, const int* in_sizes, int n_in,
                              void* d_out, int out_size, void* d_ws, size_t ws_size,
                              hipStream_t stream) {
    (void)in_sizes; (void)n_in; (void)out_size; (void)ws_size;
    const int*   ids  = (const int*)d_in[0];
    const int*   am   = (const int*)d_in[1];
    const int*   rb   = (const int*)d_in[2];
    const float* wemb = (const float*)d_in[3];
    const float* pemb = (const float*)d_in[4];
    const float* elng = (const float*)d_in[5];
    const float* elnb = (const float*)d_in[6];
    const float* Wq = (const float*)d_in[7];  const float* bq = (const float*)d_in[8];
    const float* Wk = (const float*)d_in[9];  const float* bk = (const float*)d_in[10];
    const float* Wv = (const float*)d_in[11]; const float* bv = (const float*)d_in[12];
    const float* Wo = (const float*)d_in[13]; const float* bo = (const float*)d_in[14];
    const float* ln1g = (const float*)d_in[15]; const float* ln1b = (const float*)d_in[16];
    const float* W1 = (const float*)d_in[17]; const float* b1 = (const float*)d_in[18];
    const float* W2 = (const float*)d_in[19]; const float* b2 = (const float*)d_in[20];
    const float* ln2g = (const float*)d_in[21]; const float* ln2b = (const float*)d_in[22];
    const float* Wqa = (const float*)d_in[23]; const float* bqa = (const float*)d_in[24];
    float* out = (float*)d_out;

    size_t off = 0;
    auto allocF = [&](size_t n) { float* p = (float*)((char*)d_ws + off); off += n * sizeof(float); return p; };
    auto allocH = [&](size_t n) { unsigned short* p = (unsigned short*)((char*)d_ws + off); off += n * sizeof(unsigned short); return p; };
    const size_t ND = (size_t)B_ * S_ * D_;
    const size_t NF = (size_t)B_ * S_ * F_;
    // f32 activations
    float* X  = allocF(ND);
    float* X1 = allocF(ND);
    float* Tb = allocF(ND);
    // bf16 activations
    unsigned short* Xb  = allocH(ND);
    unsigned short* X1b = allocH(ND);
    unsigned short* Qb  = allocH(ND);
    unsigned short* Kbb = allocH(ND);
    unsigned short* Vbb = allocH(ND);
    unsigned short* Abb = allocH(ND);
    unsigned short* Hbb = allocH(NF);
    // bf16 transposed weights
    unsigned short* WqT = allocH((size_t)L_ * D_ * D_);
    unsigned short* WkT = allocH((size_t)L_ * D_ * D_);
    unsigned short* WvT = allocH((size_t)L_ * D_ * D_);
    unsigned short* WoT = allocH((size_t)L_ * D_ * D_);
    unsigned short* W1T = allocH((size_t)L_ * D_ * F_);
    unsigned short* W2T = allocH((size_t)L_ * F_ * D_);
    int* idxb  = (int*)((char*)d_ws + off); off += (size_t)L_ * NB_ * 8 * sizeof(int);
    int* keepb = (int*)((char*)d_ws + off); off += (size_t)L_ * NB_ * 8 * sizeof(int);

    // one-time weight convert+transpose (re-run each call for determinism)
    dim3 gtDD(D_ / 32, D_ / 32, L_);
    dim3 gtDF(F_ / 32, D_ / 32, L_);   // W1: K=D, N=F
    dim3 gtFD(D_ / 32, F_ / 32, L_);   // W2: K=F, N=D
    conv_transpose<<<gtDD, 256, 0, stream>>>(Wq, WqT, D_, D_);
    conv_transpose<<<gtDD, 256, 0, stream>>>(Wk, WkT, D_, D_);
    conv_transpose<<<gtDD, 256, 0, stream>>>(Wv, WvT, D_, D_);
    conv_transpose<<<gtDD, 256, 0, stream>>>(Wo, WoT, D_, D_);
    conv_transpose<<<gtDF, 256, 0, stream>>>(W1, W1T, D_, F_);
    conv_transpose<<<gtFD, 256, 0, stream>>>(W2, W2T, F_, D_);

    idx_kernel<<<(L_ * NB_ + 255) / 256, 256, 0, stream>>>(rb, idxb, keepb);
    embed_ln<<<B_ * S_, 256, 0, stream>>>(ids, wemb, pemb, elng, elnb, X, Xb);

    const int M = B_ * S_;
    dim3 gD(D_ / 64, M / 64);
    dim3 gF(F_ / 64, M / 64);

    for (int l = 0; l < L_; l++) {
        const unsigned short* wqT = WqT + (size_t)l * D_ * D_;
        const unsigned short* wkT = WkT + (size_t)l * D_ * D_;
        const unsigned short* wvT = WvT + (size_t)l * D_ * D_;
        const unsigned short* woT = WoT + (size_t)l * D_ * D_;
        const unsigned short* w1T = W1T + (size_t)l * D_ * F_;
        const unsigned short* w2T = W2T + (size_t)l * F_ * D_;

        gemm_bf16<<<gD, 256, 0, stream>>>(Xb, wqT, bq + (size_t)l * D_, nullptr, Qb,  M, D_, D_, 0);
        gemm_bf16<<<gD, 256, 0, stream>>>(Xb, wkT, bk + (size_t)l * D_, nullptr, Kbb, M, D_, D_, 0);
        gemm_bf16<<<gD, 256, 0, stream>>>(Xb, wvT, bv + (size_t)l * D_, nullptr, Vbb, M, D_, D_, 0);

        sparse_attn<<<B_ * H_ * (NB_ - 2), 256, 0, stream>>>(Qb, Kbb, Vbb, am, idxb, keepb, Abb, l);
        global_attn<<<B_ * H_, 256, 0, stream>>>(Qb, Kbb, Vbb, am, Abb);

        gemm_bf16<<<gD, 256, 0, stream>>>(Abb, woT, bo + (size_t)l * D_, Tb, nullptr, M, D_, D_, 0);
        add_ln<<<B_ * S_, 256, 0, stream>>>(X, Tb, ln1g + (size_t)l * D_, ln1b + (size_t)l * D_, X1, X1b);

        gemm_bf16<<<gF, 256, 0, stream>>>(X1b, w1T, b1 + (size_t)l * F_, nullptr, Hbb, M, F_, D_, 1); // fused GeLU
        gemm_bf16<<<gD, 256, 0, stream>>>(Hbb, w2T, b2 + (size_t)l * D_, Tb, nullptr, M, D_, F_, 0);
        add_ln<<<B_ * S_, 256, 0, stream>>>(X1, Tb, ln2g + (size_t)l * D_, ln2b + (size_t)l * D_, X, Xb);
    }

    qa_proj<<<B_ * S_, 256, 0, stream>>>(X, Wqa, bqa, out);
}
